// Model_39676907883893
// MI455X (gfx1250) — matrix operand order repaired and checked by reading, not compiled, not run on hardware
//
#include <hip/hip_runtime.h>
#include <hip/hip_bf16.h>
#include <math.h>

#define Bb 2
#define Hh 32
#define SQl 2048
#define SKVl 2048
#define DD 128

#define BLOCK_Q 128          // 4 waves x 2 q-tiles x 16 rows
#define BLOCK_KV 64
#define NWAVES 4
#define NTHREADS 128

// LDS row strides (bf16 elements): rows 16B-aligned, bank-spread
#define KS 136   // K tile  [64 kv][128 d]   row = 272B (17*16)
#define VS 72    // V tile  [128 d][64 kv]   row = 144B (9*16)
#define PS 72    // P stage [32 m][64 kv]    row = 144B

typedef __attribute__((ext_vector_type(16))) __bf16 v16bf;
typedef __attribute__((ext_vector_type(8)))  __bf16 v8bf;
typedef __attribute__((ext_vector_type(8)))  float  v8f;
typedef __attribute__((ext_vector_type(4)))  unsigned short us4;

union FragA { v16bf v; v8bf h[2]; unsigned short us[16]; unsigned int u32[8]; };
union FragC { v8f v; float c[8]; };
union Half8 { v8bf v; unsigned short us[8]; unsigned int u32[4]; };

// two f32 -> packed bf16 (low = a, high = b): 2 adds + 1 v_perm_b32
__device__ __forceinline__ unsigned int f2bf2(float a, float b) {
  union { float f; unsigned int u; } xa, xb; xa.f = a; xb.f = b;
  const unsigned int ua = xa.u + 0x8000u;
  const unsigned int ub = xb.u + 0x8000u;
#if __has_builtin(__builtin_amdgcn_perm)
  return __builtin_amdgcn_perm(ub, ua, 0x07060302u);  // {ub[3:2], ua[3:2]}
#else
  return (ua >> 16) | (ub & 0xFFFF0000u);
#endif
}

__device__ __forceinline__ unsigned short f2bf(float f) {
  union { float f; unsigned int u; } x; x.f = f;
  return (unsigned short)((x.u + 0x8000u) >> 16);
}

__device__ __forceinline__ float rmax16(float v) {
  #pragma unroll
  for (int m = 1; m < 16; m <<= 1) v = fmaxf(v, __shfl_xor(v, m, 32));
  return v;
}

// async global -> LDS 16B copy (ASYNCcnt); lds_off = wave-relative LDS byte addr
__device__ __forceinline__ void async_copy16(unsigned lds_off, const float* gbase,
                                             unsigned goff) {
  asm volatile("global_load_async_to_lds_b128 %0, %1, %2"
               :: "v"(lds_off), "v"(goff), "s"(gbase) : "memory");
}

__device__ __forceinline__ void wait_async0() {
#if __has_builtin(__builtin_amdgcn_s_wait_asynccnt)
  __builtin_amdgcn_s_wait_asynccnt(0);
#else
  asm volatile("s_wait_asynccnt 0x0" ::: "memory");
#endif
}

__global__ __launch_bounds__(NTHREADS)
void fa_fwd_kernel(const float* __restrict__ Q, const float* __restrict__ K,
                   const float* __restrict__ V, const float* __restrict__ scale_p,
                   const float* __restrict__ dp_p, float* __restrict__ O) {
  __shared__ float lds_ksta[BLOCK_KV * DD];      // 32 KB fp32 staging (async dst)
  __shared__ float lds_vsta[BLOCK_KV * DD];      // 32 KB fp32 staging (async dst)
  __shared__ unsigned short lds_k[BLOCK_KV * KS];
  __shared__ unsigned short lds_vt[DD * VS];
  __shared__ unsigned short lds_p[NWAVES * 32 * PS];

  const int t     = threadIdx.x;
  const int lane  = t & 31;
  const int wave  = t >> 5;
  const int nhalf = lane >> 4;
  const int nl    = lane & 15;

  const int bh    = blockIdx.x / (SQl / BLOCK_Q);
  const int qblk  = blockIdx.x % (SQl / BLOCK_Q);
  const int qbase = qblk * BLOCK_Q + wave * 32;   // this wave: rows [qbase, qbase+32)

  // base-2 softmax: fold scale * log2(e) into Q so exp() is a bare v_exp_f32
  const float qs = scale_p[0] * 1.44269504088896340736f;
  const float dp = dp_p[0];

  const float* Qg = Q + (size_t)bh * SQl * DD;
  const float* Kg = K + (size_t)bh * SKVl * DD;
  const float* Vg = V + (size_t)bh * SKVl * DD;
  float*       Og = O + (size_t)bh * SQl * DD;

  // LDS byte offsets (generic LDS pointer: low 32 bits = LDS offset)
  const unsigned ksta_off = (unsigned)(size_t)&lds_ksta[0];
  const unsigned vsta_off = (unsigned)(size_t)&lds_vsta[0];

  // ---- async prefetch of the first K/V tile (contiguous 32 KB each) ----
  {
    const float* kbase = Kg;
    const float* vbase = Vg;
    #pragma unroll
    for (int i = 0; i < 16; ++i) {
      const unsigned goff = (unsigned)((i * NTHREADS + t) * 16);
      async_copy16(ksta_off + goff, kbase, goff);
      async_copy16(vsta_off + goff, vbase, goff);
    }
  }

  // ---- Q A-fragments: two 16x128 tiles, pre-scaled, bf16 ----
  FragA aq[2][4];
  #pragma unroll
  for (int qt = 0; qt < 2; ++qt) {
    const int qrow = qbase + qt * 16 + nl;
    const int kb   = nhalf * 8;
    #pragma unroll
    for (int kc = 0; kc < 4; ++kc) {
      #pragma unroll
      for (int half = 0; half < 2; ++half) {
        const int d0 = kc * 32 + kb + half * 16;
        const float4 f0 = *(const float4*)(Qg + (size_t)qrow * DD + d0);
        const float4 f1 = *(const float4*)(Qg + (size_t)qrow * DD + d0 + 4);
        Half8 pk;
        pk.u32[0] = f2bf2(f0.x * qs, f0.y * qs);
        pk.u32[1] = f2bf2(f0.z * qs, f0.w * qs);
        pk.u32[2] = f2bf2(f1.x * qs, f1.y * qs);
        pk.u32[3] = f2bf2(f1.z * qs, f1.w * qs);
        aq[qt][kc].h[half] = pk.v;
      }
    }
  }

  // ones B-matrix (32x16 of bf16 1.0) for WMMA row-sums
  FragA ones;
  #pragma unroll
  for (int i = 0; i < 8; ++i) ones.u32[i] = 0x3F803F80u;

  FragC oacc[2][8];
  FragC lacc[2];
  float row_max[2][8];
  #pragma unroll
  for (int qt = 0; qt < 2; ++qt) {
    #pragma unroll
    for (int n = 0; n < 8; ++n)
      #pragma unroll
      for (int r = 0; r < 8; ++r) oacc[qt][n].c[r] = 0.0f;
    #pragma unroll
    for (int r = 0; r < 8; ++r) { lacc[qt].c[r] = 0.0f; row_max[qt][r] = -INFINITY; }
  }

  unsigned short* pstage = lds_p + wave * 32 * PS;   // 32 rows per wave

  const int kvrK = t >> 1;            // K convert: row 0..63
  const int dc0K = (t & 1) * 64;      // K convert: d base 0/64
  const int kvpV = (t >> 2) * 2;      // V convert: kv pair base 0..62
  const int dc0V = (t & 3) * 32;      // V convert: d base 0/32/64/96

  for (int kv0 = 0; kv0 < SKVl; kv0 += BLOCK_KV) {
    // async tile for this iteration has landed in staging
    wait_async0();
    __syncthreads();

    // ---- convert staging fp32 -> bf16 tiles (LDS -> VALU -> LDS) ----
    {
      // K: row-major, 8B-packed stores
      #pragma unroll
      for (int i = 0; i < 16; ++i) {
        const int d = dc0K + 4 * i;
        const float4 kf = *(const float4*)&lds_ksta[kvrK * DD + d];
        union { us4 s; unsigned int u[2]; } m;
        m.u[0] = f2bf2(kf.x, kf.y);
        m.u[1] = f2bf2(kf.z, kf.w);
        *(us4*)&lds_k[kvrK * KS + d] = m.s;
      }
      // V: transposed; pack (kv, kv+1) at same d -> one b32 store
      #pragma unroll
      for (int i = 0; i < 8; ++i) {
        const int d = dc0V + 4 * i;
        const float4 f0 = *(const float4*)&lds_vsta[(kvpV + 0) * DD + d];
        const float4 f1 = *(const float4*)&lds_vsta[(kvpV + 1) * DD + d];
        *(unsigned int*)&lds_vt[(d + 0) * VS + kvpV] = f2bf2(f0.x, f1.x);
        *(unsigned int*)&lds_vt[(d + 1) * VS + kvpV] = f2bf2(f0.y, f1.y);
        *(unsigned int*)&lds_vt[(d + 2) * VS + kvpV] = f2bf2(f0.z, f1.z);
        *(unsigned int*)&lds_vt[(d + 3) * VS + kvpV] = f2bf2(f0.w, f1.w);
      }
    }
    __syncthreads();   // staging consumed + bf16 tiles complete

    // ---- kick off async copy of the NEXT tile; overlaps all WMMAs below ----
    if (kv0 + BLOCK_KV < SKVl) {
      const float* kbase = Kg + (size_t)(kv0 + BLOCK_KV) * DD;
      const float* vbase = Vg + (size_t)(kv0 + BLOCK_KV) * DD;
      #pragma unroll
      for (int i = 0; i < 16; ++i) {
        const unsigned goff = (unsigned)((i * NTHREADS + t) * 16);
        async_copy16(ksta_off + goff, kbase, goff);
        async_copy16(vsta_off + goff, vbase, goff);
      }
    }

    // ---- S = Qs K^T : each B fragment feeds both q-tiles ----
    FragC s[2][4];
    #pragma unroll
    for (int j = 0; j < 4; ++j) {
      v8f c0 = {}, c1 = {};
      #pragma unroll
      for (int kc = 0; kc < 4; ++kc) {
        FragA b;   // B 32x16: lane half h -> K(d) 8h..8h+7 / 16+8h..16+8h+7 in chunk
        const int row = 16 * j + nl;
        const int d0  = kc * 32 + nhalf * 8;
        b.h[0] = *(const v8bf*)&lds_k[row * KS + d0];
        b.h[1] = *(const v8bf*)&lds_k[row * KS + d0 + 16];
        c0 = __builtin_amdgcn_wmma_f32_16x16x32_bf16(false, aq[0][kc].v, false, b.v,
                                                     (short)0, c0, false, false);
        c1 = __builtin_amdgcn_wmma_f32_16x16x32_bf16(false, aq[1][kc].v, false, b.v,
                                                     (short)0, c1, false, false);
      }
      s[0][j].v = c0;
      s[1][j].v = c1;
    }

    // ---- online softmax (base 2); stage P as bf16 ----
    #pragma unroll
    for (int qt = 0; qt < 2; ++qt) {
      #pragma unroll
      for (int r = 0; r < 8; ++r) {
        float mx = fmaxf(fmaxf(s[qt][0].c[r], s[qt][1].c[r]),
                         fmaxf(s[qt][2].c[r], s[qt][3].c[r]));
        mx = rmax16(mx);
        const float mnew  = fmaxf(row_max[qt][r], mx);
        const float alpha = exp2f(row_max[qt][r] - mnew);
        row_max[qt][r] = mnew;
        const int mrow = qt * 16 + r + 8 * nhalf;   // C layout: VGPR r <-> rows r / r+8
        #pragma unroll
        for (int j = 0; j < 4; ++j) {
          const float p = exp2f(s[qt][j].c[r] - mnew);
          pstage[mrow * PS + 16 * j + nl] = f2bf(p);
        }
        #pragma unroll
        for (int n = 0; n < 8; ++n) oacc[qt][n].c[r] *= alpha;
        lacc[qt].c[r] *= alpha;
      }
    }

    // wave-local LDS RAW: staged P -> A-fragment reload
    asm volatile("s_wait_dscnt 0" ::: "memory");

    // ---- P A-fragments (2 q-tiles x 2 K=32 chunks) ----
    FragA pa[2][2];
    #pragma unroll
    for (int qt = 0; qt < 2; ++qt)
      #pragma unroll
      for (int c = 0; c < 2; ++c) {
        const int kb = c * 32 + nhalf * 8;
        pa[qt][c].h[0] = *(const v8bf*)&pstage[(qt * 16 + nl) * PS + kb];
        pa[qt][c].h[1] = *(const v8bf*)&pstage[(qt * 16 + nl) * PS + kb + 16];
      }

    // row-sums on the matrix unit: lacc += P @ ones
    #pragma unroll
    for (int qt = 0; qt < 2; ++qt) {
      lacc[qt].v = __builtin_amdgcn_wmma_f32_16x16x32_bf16(false, pa[qt][0].v, false, ones.v,
                                                           (short)0, lacc[qt].v, false, false);
      lacc[qt].v = __builtin_amdgcn_wmma_f32_16x16x32_bf16(false, pa[qt][1].v, false, ones.v,
                                                           (short)0, lacc[qt].v, false, false);
    }

    // ---- O += P V : each V fragment feeds both q-tiles ----
    #pragma unroll
    for (int n = 0; n < 8; ++n) {
      const int dn = n * 16 + nl;
      #pragma unroll
      for (int c = 0; c < 2; ++c) {
        FragA vb;  // B 32x16: N = d column, K = kv chunk c
        const int kb = c * 32 + nhalf * 8;
        vb.h[0] = *(const v8bf*)&lds_vt[dn * VS + kb];
        vb.h[1] = *(const v8bf*)&lds_vt[dn * VS + kb + 16];
        oacc[0][n].v = __builtin_amdgcn_wmma_f32_16x16x32_bf16(false, pa[0][c].v, false, vb.v,
                                                               (short)0, oacc[0][n].v, false, false);
        oacc[1][n].v = __builtin_amdgcn_wmma_f32_16x16x32_bf16(false, pa[1][c].v, false, vb.v,
                                                               (short)0, oacc[1][n].v, false, false);
      }
    }
  }

  // ---- epilogue: out = dropout_p * O / rowsum ----
  #pragma unroll
  for (int qt = 0; qt < 2; ++qt)
    #pragma unroll
    for (int r = 0; r < 8; ++r) {
      const float inv  = dp / lacc[qt].c[r];
      const int   mrow = qbase + qt * 16 + r + 8 * nhalf;
      #pragma unroll
      for (int n = 0; n < 8; ++n) {
        Og[(size_t)mrow * DD + n * 16 + nl] = oacc[qt][n].c[r] * inv;
      }
    }
}

extern "C" void kernel_launch(void* const* d_in, const int* in_sizes, int n_in,
                              void* d_out, int out_size, void* d_ws, size_t ws_size,
                              hipStream_t stream) {
  const float* q     = (const float*)d_in[0];
  const float* k     = (const float*)d_in[1];
  const float* v     = (const float*)d_in[2];
  const float* scale = (const float*)d_in[3];
  const float* dpp   = (const float*)d_in[4];
  float* out = (float*)d_out;

  dim3 grid(Bb * Hh * (SQl / BLOCK_Q));   // 2 * 32 * 16 = 1024 blocks
  dim3 block(NTHREADS);
  hipLaunchKernelGGL(fa_fwd_kernel, grid, block, 0, stream, q, k, v, scale, dpp, out);
}
